// MultiHeadAttention_85169201479897
// MI455X (gfx1250) — matrix operand order repaired and checked by reading, not compiled, not run on hardware
//
#include <hip/hip_runtime.h>
#include <hip/hip_bf16.h>
#include <math.h>

// ---------------------------------------------------------------------------
// Sliding-window (Longformer) attention forward for MI455X (gfx1250).
// All matmuls use V_WMMA_F32_16X16X32_BF16 (wave32). Working set ~140MB fits
// in 192MB L2, so stages communicate through global scratch at L2 bandwidth.
// GEMM tiles are staged into LDS by the Tensor Data Mover (tensor_load_to_lds,
// double-buffered, synchronized with s_wait_tensorcnt + workgroup barriers).
// ---------------------------------------------------------------------------

typedef unsigned short u16;
typedef unsigned int   u32;
typedef unsigned long long u64;
typedef __attribute__((ext_vector_type(16))) __bf16 v16bf;
typedef __attribute__((ext_vector_type(8)))  float  v8f;
typedef __attribute__((ext_vector_type(4)))  unsigned int vu4;
typedef __attribute__((ext_vector_type(8)))  int vi8;
typedef __attribute__((ext_vector_type(4)))  int vi4;

#if __has_builtin(__builtin_amdgcn_tensor_load_to_lds)
#define HAVE_TDM 1
#else
#define HAVE_TDM 0
#endif

union FragU { uint4 u[2]; v16bf v; };

__device__ __forceinline__ u16 f2bf(float f) {
  u32 u = __float_as_uint(f);
  u32 r = u + 0x7FFFu + ((u >> 16) & 1u);   // round-to-nearest-even
  return (u16)(r >> 16);
}

__device__ __forceinline__ v8f vzero8() {
  v8f z = {0.f,0.f,0.f,0.f,0.f,0.f,0.f,0.f};
  return z;
}

// --------------------------- TDM tile load ---------------------------------
// Load a 128-row x 32-col bf16 tile (row stride = stride_elems in memory) into
// LDS at byte offset lds_off, padding each 64B row with 16B (LDS stride 80B).
#if HAVE_TDM
__device__ __forceinline__ void tdm_load_tile(const u16* gptr, u32 lds_off,
                                              int stride_elems) {
  u64 ga = (u64)(uintptr_t)gptr;
  vu4 g0;
  g0[0] = 1u;                                       // count=1, user mode, no gather
  g0[1] = lds_off;                                  // LDS byte address
  g0[2] = (u32)(ga & 0xFFFFFFFFu);                  // global_addr[31:0]
  g0[3] = (u32)((ga >> 32) & 0x1FFFFFFu) | (2u << 30); // addr[56:32] | type=2
  const u32 tdim0 = 1u << 30, tdim1 = 1u << 30;     // huge: no OOB clipping
  vi8 g1;
  g1[0] = (int)((1u << 16)      // data_size = 2 bytes
              | (1u << 20)      // pad_enable
              | (3u << 22)      // pad_interval: 16 DWORDs (64B) between pads
              | (3u << 25));    // pad_amount: 4 DWORDs (16B)
  g1[1] = (int)((tdim0 & 0xFFFFu) << 16);           // tensor_dim0[15:0]
  g1[2] = (int)((tdim0 >> 16) | ((tdim1 & 0xFFFFu) << 16));
  g1[3] = (int)((tdim1 >> 16) | (32u << 16));       // tile_dim0 = 32 elems
  g1[4] = 128;                                      // tile_dim1 = 128 rows
  g1[5] = stride_elems;                             // tensor_dim0_stride lo32
  g1[6] = 0;
  g1[7] = 0;
  vi4 g2 = {0,0,0,0}, g3 = {0,0,0,0};
#if __clang_major__ >= 23
  vi8 g4 = {0,0,0,0,0,0,0,0};
  __builtin_amdgcn_tensor_load_to_lds(g0, g1, g2, g3, g4, 0);
#else
  __builtin_amdgcn_tensor_load_to_lds(g0, g1, g2, g3, 0);
#endif
}
#endif

// --------------------------- elementwise helpers ---------------------------

__global__ void k_cvt_bf16(const float* __restrict__ in, u16* __restrict__ out, int n) {
  int i = blockIdx.x * blockDim.x + threadIdx.x;
  if (i < n) out[i] = f2bf(in[i]);
}

__global__ void k_fill_zero_u16(u16* __restrict__ p, int n) {
  int i = blockIdx.x * blockDim.x + threadIdx.x;
  if (i < n) p[i] = 0;
}

// --------------------------- LDS-staged bf16 GEMM --------------------------
// C[M,N] = A[M,K] @ Bt[N,K]^T + bias ; block tile 128x128, 8 waves of 32x64.
// Tiles staged by TDM (double-buffered) when available.
#define TILE_E 5120            // 128 rows * 40 elems (80B padded rows)
#define TILE_B 10240           // tile bytes in LDS

template <bool OUT_BF16>
__global__ __launch_bounds__(256) void k_gemm(const u16* __restrict__ A,
                                              const u16* __restrict__ Bt,
                                              const float* __restrict__ bias,
                                              void* __restrict__ out,
                                              int M, int N, int K) {
  extern __shared__ __align__(16) u16 smem[];       // 2 x (A tile + B tile)
  const int t = threadIdx.x;
  const int lane = t & 31, wave = t >> 5;
  const int wm = wave & 3, wn = wave >> 2;          // 4 m-strips x 2 n-strips
  const int hl = lane >> 4, ll = lane & 15;
  const int m0 = blockIdx.y * 128, n0 = blockIdx.x * 128;
  const int iters = K >> 5;

  v8f acc[2][4];
#pragma unroll
  for (int i = 0; i < 2; ++i)
#pragma unroll
    for (int j = 0; j < 4; ++j) acc[i][j] = vzero8();

#if HAVE_TDM
  const bool issuer = (t < 32);                     // wave 0 drives the TDM
  if (issuer) {
    tdm_load_tile(&A[(size_t)m0 * K], 0, K);
    tdm_load_tile(&Bt[(size_t)n0 * K], TILE_B, K);
  }
#endif

  for (int it = 0; it < iters; ++it) {
    const int cur = it & 1;
#if HAVE_TDM
    if (issuer) {
      if (it + 1 < iters) {
        int k0n = (it + 1) << 5;
        u32 nxt = (u32)(1 - cur) * 2u * TILE_B;
        tdm_load_tile(&A[(size_t)m0 * K + k0n], nxt, K);
        tdm_load_tile(&Bt[(size_t)n0 * K + k0n], nxt + TILE_B, K);
        __builtin_amdgcn_s_wait_tensorcnt(2);       // 2 oldest (cur pair) done
      } else {
        __builtin_amdgcn_s_wait_tensorcnt(0);
      }
    }
    __syncthreads();
#else
    const int k0 = it << 5;
    __syncthreads();
#pragma unroll
    for (int c = 0; c < 2; ++c) {                   // manual staging fallback
      int ch = t * 2 + c;
      int row = ch >> 2, off = (ch & 3) * 8;
      *(uint4*)&smem[cur * 2 * TILE_E + row * 40 + off] =
          *(const uint4*)&A[(size_t)(m0 + row) * K + k0 + off];
      *(uint4*)&smem[cur * 2 * TILE_E + TILE_E + row * 40 + off] =
          *(const uint4*)&Bt[(size_t)(n0 + row) * K + k0 + off];
    }
    __syncthreads();
#endif
    const u16* sA = smem + (size_t)cur * 2 * TILE_E;
    const u16* sB = sA + TILE_E;

    v16bf af[2], bfr[4];
#pragma unroll
    for (int mt = 0; mt < 2; ++mt) {                // A frag: row ll, K=hl*8../+16
      int r = 32 * wm + 16 * mt + ll;
      int kb = hl * 8;
      FragU f;
      f.u[0] = *(const uint4*)&sA[r * 40 + kb];
      f.u[1] = *(const uint4*)&sA[r * 40 + kb + 16];
      af[mt] = f.v;
    }
#pragma unroll
    for (int nt = 0; nt < 4; ++nt) {                // B frag: col ll, K=hl*8../+16
      int r = 64 * wn + 16 * nt + ll;
      int kb = hl * 8;
      FragU f;
      f.u[0] = *(const uint4*)&sB[r * 40 + kb];
      f.u[1] = *(const uint4*)&sB[r * 40 + kb + 16];
      bfr[nt] = f.v;
    }
#pragma unroll
    for (int mt = 0; mt < 2; ++mt)
#pragma unroll
      for (int nt = 0; nt < 4; ++nt)
        acc[mt][nt] = __builtin_amdgcn_wmma_f32_16x16x32_bf16(
            false, af[mt], false, bfr[nt], (short)0, acc[mt][nt], false, false);
    __syncthreads();                                // reads done before rewrite
  }

#pragma unroll
  for (int mt = 0; mt < 2; ++mt)
#pragma unroll
    for (int nt = 0; nt < 4; ++nt) {
      int col = n0 + 64 * wn + 16 * nt + ll;
      float bv = bias[col];
#pragma unroll
      for (int r = 0; r < 8; ++r) {
        int row = m0 + 32 * wm + 16 * mt + r + 8 * hl;
        float v = acc[mt][nt][r] + bv;
        if (OUT_BF16) ((u16*)out)[(size_t)row * N + col] = f2bf(v);
        else          ((float*)out)[(size_t)row * N + col] = v;
      }
    }
}

// ---------------- K/V scatter into zero-padded buffers ---------------------
// kpad[bh][128 + s][d]  (rows 0..127 and 2176..2303 stay zero)
// vt  [bh][d][128 + s]  (cols 0..127 and 2176..2303 stay zero)
__global__ void k_scatter_kv(const u16* __restrict__ qkv,
                             u16* __restrict__ kpad, u16* __restrict__ vt) {
  int idx = blockIdx.x * blockDim.x + threadIdx.x;
  if (idx >= 16 * 2048 * 128) return;
  int d  = idx & 127;
  int s  = (idx >> 7) & 2047;
  int bh = idx >> 18;
  int b = bh >> 3, h = bh & 7;
  const u16* base = &qkv[((size_t)(b * 2048 + s)) * 3072 + h * 384];
  kpad[((size_t)bh * 2304 + 128 + s) * 128 + d] = base[128 + d];  // K
  vt[((size_t)(bh * 128 + d)) * 2304 + 128 + s] = base[256 + d];  // V^T
}

// ------------------------------ band scores --------------------------------
#define S_LEN 2048
#define WINW  384

__global__ __launch_bounds__(256) void k_scores(const u16* __restrict__ qkv,
                                                const u16* __restrict__ kpad,
                                                const int* __restrict__ pmask,
                                                float* __restrict__ sc_out) {
  const int bh = blockIdx.x;                 // b*H + h
  const int qc = blockIdx.y;
  const int h  = bh & 7;
  const int q0 = qc * 128;
  const int t = threadIdx.x, lane = t & 31, wave = t >> 5;
  const int hl = lane >> 4, ll = lane & 15;
  const int mrow = q0 + wave * 16;           // this wave's 16 query rows
  const size_t qcol = (size_t)h * 384;       // q slab within head

  v8f acc[24];
#pragma unroll
  for (int i = 0; i < 24; ++i) acc[i] = vzero8();

  for (int ks = 0; ks < 4; ++ks) {           // head_dim 128 = 4 x K32
    int ar = mrow + ll;
    int kb = ks * 32 + hl * 8;
    FragU fa;
    fa.u[0] = *(const uint4*)&qkv[(size_t)ar * 3072 + qcol + kb];
    fa.u[1] = *(const uint4*)&qkv[(size_t)ar * 3072 + qcol + kb + 16];
    v16bf aq = fa.v;
#pragma unroll
    for (int nt = 0; nt < 24; ++nt) {
      int krow = q0 + nt * 16 + ll;          // padded row = key kk + 128
      int kb2  = ks * 32 + hl * 8;
      FragU fb;
      fb.u[0] = *(const uint4*)&kpad[((size_t)bh * 2304 + krow) * 128 + kb2];
      fb.u[1] = *(const uint4*)&kpad[((size_t)bh * 2304 + krow) * 128 + kb2 + 16];
      acc[nt] = __builtin_amdgcn_wmma_f32_16x16x32_bf16(
          false, aq, false, fb.v, (short)0, acc[nt], false, false);
    }
  }

  const float scale = 0.08838834764831845f;  // 1/sqrt(128)
  const int mb = bh & 1;                     // reference head-major mask quirk
#pragma unroll
  for (int nt = 0; nt < 24; ++nt) {
    int j  = nt * 16 + ll;
    int kk = q0 - 128 + j;
    bool inr = (kk >= 0) && (kk < S_LEN);
    float padadd = 0.f;
    if (inr && pmask[mb * S_LEN + kk] == 0) padadd = -1e30f;
#pragma unroll
    for (int r = 0; r < 8; ++r) {
      int i = wave * 16 + r + 8 * hl;
      bool band = (j >= i) && (j <= i + 256);   // |s - kk| <= 128
      float v = (band && inr) ? (acc[nt][r] * scale + padadd) : -3.0e38f;
      sc_out[((size_t)bh * S_LEN + q0 + i) * WINW + j] = v;
    }
  }
}

// ------------------------------- softmax -----------------------------------
__global__ void k_softmax(const float* __restrict__ sc, u16* __restrict__ probs) {
  int row = blockIdx.x * blockDim.x + threadIdx.x;   // 16*2048 rows
  const float* p = sc + (size_t)row * WINW;
  u16* o = probs + (size_t)row * WINW;
  float m = -3.4e38f;
  for (int j = 0; j < WINW; ++j) m = fmaxf(m, p[j]);
  float sum = 0.f;
  for (int j = 0; j < WINW; ++j) sum += __expf(p[j] - m);
  float inv = 1.f / sum;
  for (int j = 0; j < WINW; ++j) o[j] = f2bf(__expf(p[j] - m) * inv);
}

// ------------------------------- P @ V -------------------------------------
__global__ __launch_bounds__(256) void k_pav(const u16* __restrict__ probs,
                                             const u16* __restrict__ vt,
                                             u16* __restrict__ valb) {
  const int bh = blockIdx.x, qc = blockIdx.y;
  const int b = bh >> 3, h = bh & 7;
  const int q0 = qc * 128;
  const int t = threadIdx.x, lane = t & 31, wave = t >> 5;
  const int hl = lane >> 4, ll = lane & 15;

  v8f acc[8];
#pragma unroll
  for (int i = 0; i < 8; ++i) acc[i] = vzero8();

  for (int ks = 0; ks < 12; ++ks) {          // 384-wide window = 12 x K32
    int ar = bh * 2048 + q0 + wave * 16 + ll;
    int jb = ks * 32 + hl * 8;
    FragU fa;
    fa.u[0] = *(const uint4*)&probs[(size_t)ar * WINW + jb];
    fa.u[1] = *(const uint4*)&probs[(size_t)ar * WINW + jb + 16];
#pragma unroll
    for (int nt = 0; nt < 8; ++nt) {
      int d   = nt * 16 + ll;
      int col = q0 + ks * 32 + hl * 8;       // padded index kk+128 = q0 + j
      FragU fb;
      fb.u[0] = *(const uint4*)&vt[(size_t)(bh * 128 + d) * 2304 + col];
      fb.u[1] = *(const uint4*)&vt[(size_t)(bh * 128 + d) * 2304 + col + 16];
      acc[nt] = __builtin_amdgcn_wmma_f32_16x16x32_bf16(
          false, fa.v, false, fb.v, (short)0, acc[nt], false, false);
    }
  }

#pragma unroll
  for (int nt = 0; nt < 8; ++nt) {
    int col = h * 128 + nt * 16 + ll;
#pragma unroll
    for (int r = 0; r < 8; ++r) {
      int row = b * 2048 + q0 + wave * 16 + r + 8 * hl;
      valb[(size_t)row * 1024 + col] = f2bf(acc[nt][r]);
    }
  }
}

// ------------------------------ launcher -----------------------------------
extern "C" void kernel_launch(void* const* d_in, const int* in_sizes, int n_in,
                              void* d_out, int out_size, void* d_ws, size_t ws_size,
                              hipStream_t stream) {
  const float* x     = (const float*)d_in[0];
  const int*   pmask = (const int*)d_in[1];
  const float* Wqkv  = (const float*)d_in[2];
  const float* bqkv  = (const float*)d_in[3];
  const float* Wo    = (const float*)d_in[4];
  const float* bo    = (const float*)d_in[5];

  char* ws = (char*)d_ws;
  size_t off = 0;
  auto alloc = [&](size_t bytes) -> char* {
    char* p = ws + off;
    off += (bytes + 255) & ~(size_t)255;
    return p;
  };
  u16*   xb    = (u16*)alloc((size_t)4096 * 1024 * 2);
  u16*   wqkvb = (u16*)alloc((size_t)3072 * 1024 * 2);
  u16*   wob   = (u16*)alloc((size_t)1024 * 1024 * 2);
  u16*   qkvb  = (u16*)alloc((size_t)4096 * 3072 * 2);
  float* scf   = (float*)alloc((size_t)16 * 2048 * 384 * 4);
  u16*   probs = (u16*)alloc((size_t)16 * 2048 * 384 * 2);
  u16*   kpad  = (u16*)alloc((size_t)16 * 2304 * 128 * 2);
  u16*   vt    = (u16*)alloc((size_t)16 * 128 * 2304 * 2);
  u16*   valb  = (u16*)alloc((size_t)4096 * 1024 * 2);

  k_cvt_bf16<<<(4096 * 1024 + 255) / 256, 256, 0, stream>>>(x, xb, 4096 * 1024);
  k_cvt_bf16<<<(3072 * 1024 + 255) / 256, 256, 0, stream>>>(Wqkv, wqkvb, 3072 * 1024);
  k_cvt_bf16<<<(1024 * 1024 + 255) / 256, 256, 0, stream>>>(Wo, wob, 1024 * 1024);

  // QKV projection: [4096,1024] @ [1024,3072] -> bf16 (TDM-staged tiles)
  k_gemm<true><<<dim3(24, 32), 256, 4 * TILE_B, stream>>>(xb, wqkvb, bqkv, qkvb,
                                                          4096, 3072, 1024);

  k_fill_zero_u16<<<(16 * 2304 * 128 + 255) / 256, 256, 0, stream>>>(kpad, 16 * 2304 * 128);
  k_fill_zero_u16<<<(16 * 128 * 2304 + 255) / 256, 256, 0, stream>>>(vt, 16 * 128 * 2304);
  k_scatter_kv<<<(16 * 2048 * 128 + 255) / 256, 256, 0, stream>>>(qkvb, kpad, vt);

  k_scores<<<dim3(16, 16), 256, 0, stream>>>(qkvb, kpad, pmask, scf);
  k_softmax<<<(16 * 2048) / 256, 256, 0, stream>>>(scf, probs);
  k_pav<<<dim3(16, 16), 256, 0, stream>>>(probs, vt, valb);

  // Output projection: [4096,1024] @ [1024,1024] + bo -> f32 d_out
  k_gemm<false><<<dim3(8, 32), 256, 4 * TILE_B, stream>>>(valb, wob, bo, d_out,
                                                          4096, 1024, 1024);
}
